// ContextAttention_17824114278929
// MI455X (gfx1250) — matrix operand order repaired and checked by reading, not compiled, not run on hardware
//
#include <hip/hip_runtime.h>
#include <hip/hip_bf16.h>

#define IN_F   512
#define EMB    256
#define NHEADS 4
#define DH     64
#define SEQ    512
#define BATCH  64
#define MTOT   (BATCH * SEQ)   // 32768 rows

typedef _Float16 h16 __attribute__((ext_vector_type(16)));
typedef _Float16 h8  __attribute__((ext_vector_type(8)));
typedef float    f8  __attribute__((ext_vector_type(8)));
typedef float    f4  __attribute__((ext_vector_type(4)));

static __device__ inline h16 pack16(h8 lo, h8 hi) {
  h16 r;
#pragma unroll
  for (int i = 0; i < 8; ++i) { r[i] = lo[i]; r[i + 8] = hi[i]; }
  return r;
}

static __device__ inline f8 wmma_f16(h16 a, h16 b, f8 c) {
  // D = A(16x32 f16) * B(32x16 f16) + C(16x16 f32)
  return __builtin_amdgcn_wmma_f32_16x16x32_f16(
      /*neg_a=*/false, a, /*neg_b=*/false, b,
      /*c_mod=*/(short)0, c, /*reuse_a=*/false, /*reuse_b=*/false);
}

// ---------------------------------------------------------------------------
// Prep: f32 -> f16 conversion of activations (8 elements per thread, b128 IO)
// ---------------------------------------------------------------------------
__global__ void cvt_x_kernel(const f4* __restrict__ X4, _Float16* __restrict__ Xh) {
  const int i = blockIdx.x * blockDim.x + threadIdx.x;   // one per 8 floats
  f4 a = X4[2 * i];
  f4 b = X4[2 * i + 1];
  h8 o;
#pragma unroll
  for (int j = 0; j < 4; ++j) { o[j] = (_Float16)a[j]; o[j + 4] = (_Float16)b[j]; }
  *(h8*)(Xh + 8 * (size_t)i) = o;
}

// Prep: weights transposed + converted to f16.
// WqT/WkT/WvT: [EMB][IN_F] from W[IN_F][EMB];  WoT: [IN_F][EMB] from Wo[EMB][IN_F]
__global__ void prep_w_kernel(const float* __restrict__ Wq, const float* __restrict__ Wk,
                              const float* __restrict__ Wv, const float* __restrict__ Wo,
                              _Float16* __restrict__ WqT, _Float16* __restrict__ WkT,
                              _Float16* __restrict__ WvT, _Float16* __restrict__ WoT) {
  int which = blockIdx.y;
  int i = blockIdx.x * blockDim.x + threadIdx.x;   // 0..131071
  if (which < 3) {
    const float* W = (which == 0) ? Wq : (which == 1) ? Wk : Wv;
    _Float16*   Wt = (which == 0) ? WqT : (which == 1) ? WkT : WvT;
    int n = i >> 9;          // 0..255
    int k = i & 511;         // 0..511
    Wt[i] = (_Float16)W[(size_t)k * EMB + n];
  } else {
    int n = i >> 8;          // 0..511
    int k = i & 255;         // 0..255
    WoT[i] = (_Float16)Wo[(size_t)k * IN_F + n];
  }
}

// ---------------------------------------------------------------------------
// Stage 1: QKV projection.  grid = (MTOT/32, 3).  8 waves: 2 m-subtiles x 4
// n-subtiles of 64.  Explicitly double-buffered k-loop (fully unrolled).
// ---------------------------------------------------------------------------
__global__ __launch_bounds__(256) void qkv_gemm_kernel(
    const _Float16* __restrict__ Xh,
    const _Float16* __restrict__ WqT, const _Float16* __restrict__ WkT,
    const _Float16* __restrict__ WvT,
    const float* __restrict__ bq, const float* __restrict__ bk,
    const float* __restrict__ bv,
    _Float16* __restrict__ Qh, _Float16* __restrict__ Kh, _Float16* __restrict__ Vt) {
  const int which = blockIdx.y;
  const _Float16* Wt = (which == 0) ? WqT : (which == 1) ? WkT : WvT;
  const float* bias = (which == 0) ? bq : (which == 1) ? bk : bv;

  const int lane = threadIdx.x & 31;
  const int wave = threadIdx.x >> 5;
  const int g = lane >> 4;          // lane group (0/1)
  const int ln = lane & 15;
  const int m0 = blockIdx.x * 32 + (wave >> 2) * 16;
  const int n0 = (wave & 3) * 64;

  const _Float16* aptr = Xh + (size_t)(m0 + ln) * IN_F + g * 8;
  const _Float16* bptr[4];
#pragma unroll
  for (int f = 0; f < 4; ++f)
    bptr[f] = Wt + (size_t)(n0 + f * 16 + ln) * IN_F + g * 8;

  f8 acc[4] = {};
  h16 a_cur = pack16(*(const h8*)(aptr), *(const h8*)(aptr + 16));
  h16 b_cur[4];
#pragma unroll
  for (int f = 0; f < 4; ++f) b_cur[f] = pack16(*(const h8*)(bptr[f]), *(const h8*)(bptr[f] + 16));

#pragma unroll
  for (int ks = 0; ks < 16; ++ks) {
    h16 a_nxt;
    h16 b_nxt[4];
    if (ks < 15) {
      const int k1 = (ks + 1) * 32;
      a_nxt = pack16(*(const h8*)(aptr + k1), *(const h8*)(aptr + k1 + 16));
#pragma unroll
      for (int f = 0; f < 4; ++f) b_nxt[f] = pack16(*(const h8*)(bptr[f] + k1), *(const h8*)(bptr[f] + k1 + 16));
    }
#pragma unroll
    for (int f = 0; f < 4; ++f) acc[f] = wmma_f16(a_cur, b_cur[f], acc[f]);
    if (ks < 15) {
      a_cur = a_nxt;
#pragma unroll
      for (int f = 0; f < 4; ++f) b_cur[f] = b_nxt[f];
    }
  }

  const float qscale = (which == 0) ? 0.0625f : 1.0f;   // 1/sqrt(256)
#pragma unroll
  for (int f = 0; f < 4; ++f) {
    const int n = n0 + f * 16 + ln;
    const float bb = bias[n];
#pragma unroll
    for (int r = 0; r < 8; ++r) {
      const int m = m0 + r + 8 * g;
      float v = (acc[f][r] + bb) * qscale;
      if (which == 2) {
        int bi = m >> 9, s = m & 511;
        int hh = n >> 6, d = n & 63;
        Vt[(((size_t)bi * NHEADS + hh) * DH + d) * SEQ + s] = (_Float16)v;
      } else if (which == 0) {
        Qh[(size_t)m * EMB + n] = (_Float16)v;
      } else {
        Kh[(size_t)m * EMB + n] = (_Float16)v;
      }
    }
  }
}

// ---------------------------------------------------------------------------
// Stage 2: attention.  One workgroup per (b, h, 32-row s-block).
// ---------------------------------------------------------------------------
#define SSTRIDE 514   // f32 score row stride: 514 % 64 == 2 -> conflict-free row scans
#define PSTRIDE 520   // f16 prob row stride (16B aligned rows, conflict-free frag reads)

__global__ __launch_bounds__(256) void attn_kernel(
    const _Float16* __restrict__ Qh, const _Float16* __restrict__ Kh,
    const _Float16* __restrict__ Vt, _Float16* __restrict__ Ctx,
    const float* __restrict__ gp, const float* __restrict__ tp) {
  __shared__ float sc[32 * SSTRIDE];
  __shared__ __align__(16) _Float16 pr[32 * PSTRIDE];
  __shared__ float redA[256];
  __shared__ float redB[256];

  const int nblk = SEQ / 32;                 // 16
  const int bh = blockIdx.x / nblk;
  const int sblk = blockIdx.x % nblk;
  const int b = bh >> 2, h = bh & 3;
  const int s0 = sblk * 32;

  const float gamma = gp[0];
  const float theta = tp[0];

  const int lane = threadIdx.x & 31;
  const int wave = threadIdx.x >> 5;
  const int g = lane >> 4, ln = lane & 15;

  // Preload Q fragments: 2 m-subtiles x 2 dh-halves
  h16 aq[2][2];
#pragma unroll
  for (int ms = 0; ms < 2; ++ms) {
    const int s = s0 + ms * 16 + ln;
    const _Float16* qp = Qh + (size_t)(b * SEQ + s) * EMB + h * DH;
#pragma unroll
    for (int di = 0; di < 2; ++di) {
      h8 lo = *(const h8*)(qp + di * 32 + g * 8);
      h8 hi = *(const h8*)(qp + di * 32 + g * 8 + 16);
      aq[ms][di] = pack16(lo, hi);
    }
  }

  // ---- Phase 1: scores (QK^T scaled) + positional bias into LDS ----
  for (int tt = wave; tt < 64; tt += 8) {
    const int ms = tt & 1;
    const int t0 = (tt >> 1) * 16;
    const int t = t0 + ln;
    const _Float16* kp = Kh + (size_t)(b * SEQ + t) * EMB + h * DH + g * 8;
    h16 b0 = pack16(*(const h8*)(kp), *(const h8*)(kp + 16));
    h16 b1 = pack16(*(const h8*)(kp + 32), *(const h8*)(kp + 48));
    f8 c = {};
    c = wmma_f16(aq[ms][0], b0, c);
    c = wmma_f16(aq[ms][1], b1, c);
#pragma unroll
    for (int r = 0; r < 8; ++r) {
      const int sl = ms * 16 + r + 8 * g;
      const float diff = (float)(s0 + sl - t);
      const float pe = expf(-fabsf(gamma * diff * diff - theta));
      sc[sl * SSTRIDE + t] = c[r] + pe;
    }
  }
  __syncthreads();

  // ---- Phase 2: dual softmax (full row + |s-t|<=32 band) ----
  // row = tid & 31 -> each lane of a wave owns a distinct row (no LDS conflicts)
  const int row = threadIdx.x & 31;
  const int seg = threadIdx.x >> 5;          // 8 segments of 64 columns
  const int c0 = seg * 64;
  const int sg = s0 + row;
  float* rp = sc + row * SSTRIDE;

  float p1 = -3.4e38f, p2 = -3.4e38f;
  for (int i = 0; i < 64; ++i) {
    const int t = c0 + i;
    const float x = rp[t];
    p1 = fmaxf(p1, x);
    int dd = sg - t; dd = (dd < 0) ? -dd : dd;
    if (dd <= 32) p2 = fmaxf(p2, x);
  }
  redA[threadIdx.x] = p1; redB[threadIdx.x] = p2;
  __syncthreads();
  float m1 = -3.4e38f, m2 = -3.4e38f;
#pragma unroll
  for (int i = 0; i < 8; ++i) {
    m1 = fmaxf(m1, redA[i * 32 + row]);
    m2 = fmaxf(m2, redB[i * 32 + row]);
  }
  __syncthreads();
  // Sum pass: cache e1 = exp(x - m1) back into the score row (this thread owns
  // its 64-column segment exclusively, so the overwrite is race-free).
  float su1 = 0.f, su2 = 0.f;
  for (int i = 0; i < 64; ++i) {
    const int t = c0 + i;
    const float x = rp[t];
    const float e1 = expf(x - m1);
    su1 += e1;
    int dd = sg - t; dd = (dd < 0) ? -dd : dd;
    if (dd <= 32) su2 += expf(x - m2);
    rp[t] = e1;
  }
  redA[threadIdx.x] = su1; redB[threadIdx.x] = su2;
  __syncthreads();
  float s1 = 0.f, s2 = 0.f;
#pragma unroll
  for (int i = 0; i < 8; ++i) { s1 += redA[i * 32 + row]; s2 += redB[i * 32 + row]; }
  const float i1 = 0.5f / s1;
  const float i2 = 0.5f / s2;
  // recover x for band columns: x = log(e1) + m1  -> instead recompute from e1:
  // e2 = exp(x - m2) = e1 * exp(m1 - m2)
  const float bandscale = expf(m1 - m2) * i2;
  for (int i = 0; i < 64; ++i) {
    const int t = c0 + i;
    const float e1 = rp[t];
    float p = e1 * i1;
    int dd = sg - t; dd = (dd < 0) ? -dd : dd;
    if (dd <= 32) p += e1 * bandscale;
    pr[row * PSTRIDE + t] = (_Float16)p;
  }
  __syncthreads();

  // ---- Phase 3: out = probs @ V  (A from LDS, B from transposed V) ----
  const int ms3 = wave >> 2;
  const int n0 = (wave & 3) * 16;
  f8 acc = {};
  const _Float16* prp = pr + (size_t)(ms3 * 16 + ln) * PSTRIDE;
  const _Float16* vp = Vt + ((size_t)bh * DH + n0 + ln) * SEQ;
#pragma unroll
  for (int ks = 0; ks < 16; ++ks) {
    const int k0 = ks * 32;
    h8 lo = *(const h8*)(prp + k0 + g * 8);
    h8 hi = *(const h8*)(prp + k0 + g * 8 + 16);
    h16 a = pack16(lo, hi);
    h16 bfr = pack16(*(const h8*)(vp + k0 + g * 8), *(const h8*)(vp + k0 + g * 8 + 16));
    acc = wmma_f16(a, bfr, acc);
  }
#pragma unroll
  for (int r = 0; r < 8; ++r) {
    const int sl = ms3 * 16 + r + 8 * g;
    const int d = h * DH + n0 + ln;
    Ctx[(size_t)(b * SEQ + s0 + sl) * EMB + d] = (_Float16)acc[r];
  }
}

// ---------------------------------------------------------------------------
// Stage 3: output projection.  grid = MTOT/32.  Wave: 16 rows x 128 cols.
// All 8 B fragments are loaded as one clause per k-step before the WMMAs.
// ---------------------------------------------------------------------------
__global__ __launch_bounds__(256) void out_gemm_kernel(
    const _Float16* __restrict__ Ctx, const _Float16* __restrict__ WoT,
    const float* __restrict__ bout, float* __restrict__ out) {
  const int lane = threadIdx.x & 31;
  const int wave = threadIdx.x >> 5;
  const int g = lane >> 4, ln = lane & 15;
  const int m0 = blockIdx.x * 32 + (wave >> 2) * 16;
  const int nb = (wave & 3) * 128;

  const _Float16* ap = Ctx + (size_t)(m0 + ln) * EMB + g * 8;
  const _Float16* bp = WoT + (size_t)(nb + ln) * EMB + g * 8;

  f8 acc[8] = {};
  h16 a_cur = pack16(*(const h8*)(ap), *(const h8*)(ap + 16));
#pragma unroll
  for (int ks = 0; ks < 8; ++ks) {
    const int k0 = ks * 32;
    h16 bfr[8];
#pragma unroll
    for (int f = 0; f < 8; ++f)
      bfr[f] = pack16(*(const h8*)(bp + (size_t)f * 16 * EMB + k0), *(const h8*)(bp + (size_t)f * 16 * EMB + k0 + 16));
    h16 a_nxt;
    if (ks < 7) {
      const int k1 = k0 + 32;
      a_nxt = pack16(*(const h8*)(ap + k1), *(const h8*)(ap + k1 + 16));
    }
#pragma unroll
    for (int f = 0; f < 8; ++f) acc[f] = wmma_f16(a_cur, bfr[f], acc[f]);
    if (ks < 7) a_cur = a_nxt;
  }
#pragma unroll
  for (int f = 0; f < 8; ++f) {
    const int n = nb + f * 16 + ln;
    const float bo = bout[n];
#pragma unroll
    for (int r = 0; r < 8; ++r) {
      out[(size_t)(m0 + r + 8 * g) * IN_F + n] = acc[f][r] + bo;
    }
  }
}

// ---------------------------------------------------------------------------
extern "C" void kernel_launch(void* const* d_in, const int* in_sizes, int n_in,
                              void* d_out, int out_size, void* d_ws, size_t ws_size,
                              hipStream_t stream) {
  const float* X     = (const float*)d_in[0];
  const float* Wq    = (const float*)d_in[1];
  const float* bq    = (const float*)d_in[2];
  const float* Wk    = (const float*)d_in[3];
  const float* bk    = (const float*)d_in[4];
  const float* Wv    = (const float*)d_in[5];
  const float* bv    = (const float*)d_in[6];
  const float* gamma = (const float*)d_in[7];
  const float* theta = (const float*)d_in[8];
  const float* Wo    = (const float*)d_in[9];
  const float* bo    = (const float*)d_in[10];
  float* out = (float*)d_out;

  char* ws = (char*)d_ws;
  size_t off = 0;
  _Float16* Xh  = (_Float16*)(ws + off); off += (size_t)MTOT * IN_F * 2;   // 32 MB
  _Float16* Qh  = (_Float16*)(ws + off); off += (size_t)MTOT * EMB * 2;    // 16 MB
  _Float16* Kh  = (_Float16*)(ws + off); off += (size_t)MTOT * EMB * 2;    // 16 MB
  _Float16* Vt  = (_Float16*)(ws + off); off += (size_t)MTOT * EMB * 2;    // 16 MB
  _Float16* Ctx = (_Float16*)(ws + off); off += (size_t)MTOT * EMB * 2;    // 16 MB
  _Float16* WqT = (_Float16*)(ws + off); off += (size_t)EMB * IN_F * 2;
  _Float16* WkT = (_Float16*)(ws + off); off += (size_t)EMB * IN_F * 2;
  _Float16* WvT = (_Float16*)(ws + off); off += (size_t)EMB * IN_F * 2;
  _Float16* WoT = (_Float16*)(ws + off); off += (size_t)IN_F * EMB * 2;

  const int nvec = (MTOT * IN_F) / 8;   // 2,097,152 threads, 8 floats each
  cvt_x_kernel<<<nvec / 256, 256, 0, stream>>>((const f4*)X, Xh);
  prep_w_kernel<<<dim3(512, 4), 256, 0, stream>>>(Wq, Wk, Wv, Wo, WqT, WkT, WvT, WoT);
  qkv_gemm_kernel<<<dim3(MTOT / 32, 3), 256, 0, stream>>>(Xh, WqT, WkT, WvT,
                                                          bq, bk, bv, Qh, Kh, Vt);
  attn_kernel<<<BATCH * NHEADS * (SEQ / 32), 256, 0, stream>>>(Qh, Kh, Vt, Ctx,
                                                               gamma, theta);
  out_gemm_kernel<<<MTOT / 32, 256, 0, stream>>>(Ctx, WoT, bo, out);
}
